// KGHRec_32117765440057
// MI455X (gfx1250) — matrix operand order repaired and checked by reading, not compiled, not run on hardware
//
#include <hip/hip_runtime.h>
#include <hip/hip_bf16.h>

typedef __attribute__((ext_vector_type(16))) _Float16 v16h;
typedef __attribute__((ext_vector_type(8)))  _Float16 v8h;
typedef __attribute__((ext_vector_type(8)))  float    v8f;

#define DD    128
#define NENT  100000
#define NHE   20000

// ---------------- zero fill ----------------
__global__ void zero_f32(float* __restrict__ p, long n) {
    long i = (long)blockIdx.x * blockDim.x + threadIdx.x;
    long stride = (long)gridDim.x * blockDim.x;
    for (; i < n; i += stride) p[i] = 0.0f;
}

// ---------------- COO SpMM: one wave (32 lanes) per nonzero ----------------
// out[row] += val * x[col]   (128 floats; lane handles 4 via float4 + 4 atomics)
__global__ __launch_bounds__(256) void spmm_atomic(
    const int* __restrict__ rows, const int* __restrict__ cols,
    const float* __restrict__ vals, const float* __restrict__ x,
    float* __restrict__ out, int nnz) {
    int wid  = blockIdx.x * 8 + (threadIdx.x >> 5);
    if (wid >= nnz) return;
    int lane = threadIdx.x & 31;
    int r = rows[wid];
    int c = cols[wid];
    float v = vals[wid];
    const float4 xv = ((const float4*)(x + (size_t)c * DD))[lane];
    float* op = out + (size_t)r * DD + lane * 4;
    atomicAdd(op + 0, v * xv.x);
    atomicAdd(op + 1, v * xv.y);
    atomicAdd(op + 2, v * xv.z);
    atomicAdd(op + 3, v * xv.w);
}

// XOR-swizzled W LDS address (f16): param k is the row (W's n), param n the column (W's k).
// Granule = 8 halves (16B). Swizzle group = (n>>3) ^ (k&15)  -> for B loads
// (row k = nt*16 + (lane&15), so k&15 == lane&15) lanes 0..15 cover all 16
// granules = all 64 banks exactly once; lanes 16..31 repeat -> 2-way (optimal).
__device__ __forceinline__ int wswz(int k, int n) {
    return k * DD + ((((n >> 3) ^ (k & 15)) & 15) << 3) + (n & 7);
}

// ---------------- fused bi-interaction GEMM (single pass, WMMA f16->f32) ----
// out = leaky((ego+side)@W1^T + b1) + leaky((ego*side)@W2^T + b2)
// One wave owns one 16-row tile: A-fragments (a = e+s, m = e*s; 4 K-chunks)
// built once from global (L2-resident) and reused from registers across all
// 8 N-tiles; B-fragments streamed from swizzled LDS. No barriers in the loop.
__global__ __launch_bounds__(256) void gemm_bi_fused(
    const float* __restrict__ ego, const float* __restrict__ side,
    const float* __restrict__ W1, const float* __restrict__ b1,
    const float* __restrict__ W2, const float* __restrict__ b2,
    float* __restrict__ out, int mblocks) {
    __shared__ __attribute__((aligned(32))) _Float16 w1t[DD * DD];  // 32 KB
    __shared__ __attribute__((aligned(32))) _Float16 w2t[DD * DD];  // 32 KB

    // one-time: coalesced read of W (row-major n,k), swizzled f16 W store
    for (int idx = threadIdx.x; idx < DD * DD; idx += 256) {
        int n = idx >> 7;
        int k = idx & 127;
        int d = wswz(n, k);
        w1t[d] = (_Float16)W1[idx];
        w2t[d] = (_Float16)W2[idx];
    }
    __syncthreads();

    const int lane  = threadIdx.x & 31;
    const int arow  = lane & 15;            // A layout: row within 16-row tile
    const int agrp  = lane >> 4;            // A layout: K sub-band group
    const int r0    = (lane >> 4) * 8;      // C/D layout: row base for this lane
    const int ncol  = lane & 15;            // C/D layout: column within N tile
    const int lsw   = lane & 15;            // B swizzle lane term

    // preload biases for the 8 N-tiles this lane's column touches
    float b1v[8], b2v[8];
    #pragma unroll
    for (int nt = 0; nt < 8; nt++) {
        b1v[nt] = b1[nt * 16 + ncol];
        b2v[nt] = b2[nt * 16 + ncol];
    }

    const int gwave  = (int)((blockIdx.x * blockDim.x + threadIdx.x) >> 5);
    const int nwaves = (int)((gridDim.x * blockDim.x) >> 5);

    for (int rb = gwave; rb < mblocks; rb += nwaves) {
        const size_t rowbase = (size_t)(rb * 16 + arow) * DD;

        // prefetch next tile's gather lines (gfx1250 global_prefetch path)
        {
            int rbn = rb + nwaves;
            if (rbn < mblocks) {
                const size_t nb = (size_t)(rbn * 16 + arow) * DD + agrp * 64;
                __builtin_prefetch(ego  + nb, 0, 3);
                __builtin_prefetch(side + nb, 0, 3);
            }
        }

        // ---- build A-fragments for all 4 K-chunks: a = e+s, m = e*s ----
        v16h afr[4], mfr[4];
        #pragma unroll
        for (int t = 0; t < 4; t++) {
            const int kc = t * 32;
            const float* ep = ego  + rowbase + kc + agrp * 8;
            const float* sp = side + rowbase + kc + agrp * 8;
            float e[16], s[16];
            #pragma unroll
            for (int j = 0; j < 8; j++) {
                e[j]     = ep[j];
                e[8 + j] = ep[16 + j];
                s[j]     = sp[j];
                s[8 + j] = sp[16 + j];
            }
            #pragma unroll
            for (int j = 0; j < 16; j++) {
                afr[t][j] = (_Float16)(e[j] + s[j]);
                mfr[t][j] = (_Float16)(e[j] * s[j]);
            }
        }

        // ---- sweep the 8 N-tiles (fully unrolled), B from swizzled LDS ----
        #pragma unroll
        for (int nt = 0; nt < 8; nt++) {
            v8f c1 = {0.f, 0.f, 0.f, 0.f, 0.f, 0.f, 0.f, 0.f};
            v8f c2 = {0.f, 0.f, 0.f, 0.f, 0.f, 0.f, 0.f, 0.f};
            #pragma unroll
            for (int t = 0; t < 4; t++) {
                const int gLo = (((t * 4 + agrp)     ^ lsw) & 15) << 3;
                const int gHi = (((t * 4 + 2 + agrp) ^ lsw) & 15) << 3;
                const int kbase = (nt * 16 + ncol) * DD;
                v8h lo1 = *(const v8h*)(w1t + kbase + gLo);
                v8h hi1 = *(const v8h*)(w1t + kbase + gHi);
                v16h bA = __builtin_shufflevector(lo1, hi1,
                    0, 1, 2, 3, 4, 5, 6, 7, 8, 9, 10, 11, 12, 13, 14, 15);
                c1 = __builtin_amdgcn_wmma_f32_16x16x32_f16(
                         false, afr[t], false, bA, (short)0, c1, false, false);
                v8h lo2 = *(const v8h*)(w2t + kbase + gLo);
                v8h hi2 = *(const v8h*)(w2t + kbase + gHi);
                v16h bB = __builtin_shufflevector(lo2, hi2,
                    0, 1, 2, 3, 4, 5, 6, 7, 8, 9, 10, 11, 12, 13, 14, 15);
                c2 = __builtin_amdgcn_wmma_f32_16x16x32_f16(
                         false, mfr[t], false, bB, (short)0, c2, false, false);
            }
            // epilogue: bias + LeakyReLU(0.01), sum both branches, store
            const int n = nt * 16 + ncol;
            #pragma unroll
            for (int i = 0; i < 8; i++) {
                float v1 = c1[i] + b1v[nt];
                v1 = v1 > 0.0f ? v1 : 0.01f * v1;
                float v2 = c2[i] + b2v[nt];
                v2 = v2 > 0.0f ? v2 : 0.01f * v2;
                out[(size_t)(rb * 16 + r0 + i) * DD + n] = v1 + v2;
            }
        }
    }
}

extern "C" void kernel_launch(void* const* d_in, const int* in_sizes, int n_in,
                              void* d_out, int out_size, void* d_ws, size_t ws_size,
                              hipStream_t stream) {
    const float* ego     = (const float*)d_in[0];
    const int*   a_rows  = (const int*)  d_in[1];
    const int*   a_cols  = (const int*)  d_in[2];
    const float* a_vals  = (const float*)d_in[3];
    const int*   p1_rows = (const int*)  d_in[4];
    const int*   p1_cols = (const int*)  d_in[5];
    const float* p1_vals = (const float*)d_in[6];
    const int*   p2_rows = (const int*)  d_in[7];
    const int*   p2_cols = (const int*)  d_in[8];
    const float* p2_vals = (const float*)d_in[9];
    const int*   l1_rows = (const int*)  d_in[10];
    const int*   l1_cols = (const int*)  d_in[11];
    const float* l1_vals = (const float*)d_in[12];
    const int*   l2_rows = (const int*)  d_in[13];
    const int*   l2_cols = (const int*)  d_in[14];
    const float* l2_vals = (const float*)d_in[15];
    const float* W1      = (const float*)d_in[16];
    const float* b1      = (const float*)d_in[17];
    const float* W2      = (const float*)d_in[18];
    const float* b2      = (const float*)d_in[19];
    float* out = (float*)d_out;

    // workspace layout: side[100000*128] | h_p[20000*128] | h_l[20000*128]
    float* side = (float*)d_ws;
    float* h_p  = side + (size_t)NENT * DD;
    float* h_l  = h_p  + (size_t)NHE  * DD;
    long zn = (long)NENT * DD + 2L * NHE * DD;
    zero_f32<<<2048, 256, 0, stream>>>(side, zn);

    int nnzA  = in_sizes[1];
    int nnzP1 = in_sizes[4];
    int nnzP2 = in_sizes[7];
    int nnzL1 = in_sizes[10];
    int nnzL2 = in_sizes[13];

    // side = A_in @ ego
    spmm_atomic<<<(nnzA  + 7) / 8, 256, 0, stream>>>(a_rows,  a_cols,  a_vals,  ego, side, nnzA);
    // h_p = P1 @ ego ; side += P2 @ h_p
    spmm_atomic<<<(nnzP1 + 7) / 8, 256, 0, stream>>>(p1_rows, p1_cols, p1_vals, ego, h_p,  nnzP1);
    spmm_atomic<<<(nnzP2 + 7) / 8, 256, 0, stream>>>(p2_rows, p2_cols, p2_vals, h_p, side, nnzP2);
    // h_l = L1 @ ego ; side += L2 @ h_l
    spmm_atomic<<<(nnzL1 + 7) / 8, 256, 0, stream>>>(l1_rows, l1_cols, l1_vals, ego, h_l,  nnzL1);
    spmm_atomic<<<(nnzL2 + 7) / 8, 256, 0, stream>>>(l2_rows, l2_cols, l2_vals, h_l, side, nnzL2);

    // fused bi-interaction in ONE pass:
    // out = leaky((ego+side)@W1^T+b1) + leaky((ego*side)@W2^T+b2)
    int mblocks = NENT / 16;   // 6250, exact
    gemm_bi_fused<<<512, 256, 0, stream>>>(ego, side, W1, b1, W2, b2, out, mblocks);
}
